// LegalGraphConvLayer_867583393905
// MI455X (gfx1250) — matrix operand order repaired and checked by reading, not compiled, not run on hardware
//
#include <hip/hip_runtime.h>
#include <hip/hip_bf16.h>
#include <cstdint>
#include <cstddef>

// Problem constants (match reference)
constexpr int KD     = 768;     // feature dim D
constexpr int K2D    = 1536;    // 2*D
constexpr int NNODES = 100000;
constexpr int NEDGES = 100000;
constexpr int NTYPES = 5;

// GEMM tiling
constexpr int BM = 128;         // rows per block tile
constexpr int BN = 64;          // cols per block tile
constexpr int BK = 32;          // K per stage (one bf16 WMMA K)
constexpr int LDSS = 40;        // padded LDS row stride (elements) -> 80B, conflict-free

typedef __attribute__((ext_vector_type(16))) __bf16 v16bf;
typedef __attribute__((ext_vector_type(8)))  __bf16 v8bf;
typedef __attribute__((ext_vector_type(8)))  float  v8f;

// f32 pair -> packed bf16 dword (RNE), 32-bit ALU only.
__device__ __forceinline__ uint32_t pack2bf(float lo, float hi) {
  uint32_t ul = __builtin_bit_cast(uint32_t, lo);
  uint32_t uh = __builtin_bit_cast(uint32_t, hi);
  ul = ul + 0x7FFFu + ((ul >> 16) & 1u);             // round-to-nearest-even
  uh = uh + 0x7FFFu + ((uh >> 16) & 1u);
  return __builtin_amdgcn_perm(uh, ul, 0x07060302);  // [uh:31-16 | ul:31-16]
}
__device__ __forceinline__ v16bf cat16(v8bf lo, v8bf hi) {
  return __builtin_shufflevector(lo, hi, 0,1,2,3,4,5,6,7,8,9,10,11,12,13,14,15);
}
__device__ __forceinline__ v16bf ldfragA(const __bf16* p) {   // two 16B chunks at +0,+32B
  return cat16(*(const v8bf*)p, *(const v8bf*)(p + 16));
}
__device__ __forceinline__ v16bf ldfragB(const __bf16* p) {   // contiguous 32B
  return cat16(*(const v8bf*)p, *(const v8bf*)(p + 8));
}

// ---------------- setup kernels ----------------

// generic f32 -> bf16 (packed pair) conversion, grid-stride
__global__ void k_f32_to_bf16(const float2* __restrict__ src,
                              uint32_t* __restrict__ dst, int64_t npairs) {
  int64_t i = (int64_t)blockIdx.x * blockDim.x + threadIdx.x;
  int64_t stride = (int64_t)gridDim.x * blockDim.x;
  for (int64_t j = i; j < npairs; j += stride) {
    float2 f = src[j];
    dst[j] = pack2bf(f.x, f.y);
  }
}

__global__ void k_zero(float4* __restrict__ agg4, int* __restrict__ meta) {
  const int64_t n4 = (int64_t)NNODES * KD / 4;
  int64_t i = (int64_t)blockIdx.x * blockDim.x + threadIdx.x;
  int64_t stride = (int64_t)gridDim.x * blockDim.x;
  float4 z; z.x = 0.f; z.y = 0.f; z.z = 0.f; z.w = 0.f;
  for (int64_t j = i; j < n4; j += stride) agg4[j] = z;
  if (blockIdx.x == 0 && threadIdx.x < 16) meta[threadIdx.x] = 0;
}

// meta layout: [0..4] counts, [5..9] cursors, [10..15] offsets (offsets[5]=E)
__global__ void k_hist(const int* __restrict__ etype, int* __restrict__ meta) {
  int e = blockIdx.x * blockDim.x + threadIdx.x;
  if (e < NEDGES) atomicAdd(&meta[etype[e]], 1);
}
__global__ void k_offsets(int* __restrict__ meta) {
  if (threadIdx.x == 0) {
    int acc = 0;
    for (int t = 0; t < NTYPES; ++t) { meta[10 + t] = acc; acc += meta[t]; }
    meta[15] = acc;
  }
}
__global__ void k_scatter(const int* __restrict__ etype, int* __restrict__ meta,
                          int* __restrict__ bucket) {
  int e = blockIdx.x * blockDim.x + threadIdx.x;
  if (e < NEDGES) {
    int t = etype[e];
    int p = atomicAdd(&meta[5 + t], 1);
    bucket[meta[10 + t] + p] = e;
  }
}

// ---------------- GEMM 1: typed edge messages + scatter-add ----------------
// grid: (ceil(E/BM), D/BN, NTYPES), block: 256 (8 waves as 4M x 2N, 32x32 each)
// A (gathered bf16 node rows) and B (bf16 weights) staged as pure copies.
// Software pipelined; OOB rows clamped, discarded at epilogue (EXEC all-ones
// throughout the K-loop as WMMA requires).
__launch_bounds__(256)
__global__ void k_msg_gemm(const uint16_t* __restrict__ xbf,     // [N][D] bf16
                           const uint16_t* __restrict__ wmsg_bf, // [T][out][in] bf16
                           const float* __restrict__ bmsg,       // [T][D]
                           const int* __restrict__ eidx,         // [2][E]
                           const int* __restrict__ meta,
                           const int* __restrict__ bucket,
                           float* __restrict__ agg) {
  __shared__ __align__(16) __bf16 sA[BM * LDSS];
  __shared__ __align__(16) __bf16 sB[BN * LDSS];

  const int t        = blockIdx.z;
  const int tCount   = meta[t];
  const int tStart   = meta[10 + t];
  const int tileBase = blockIdx.x * BM;
  if (tileBase >= tCount) return;
  const int colBase  = blockIdx.y * BN;

  const int tid  = threadIdx.x;
  const int lane = tid & 31;
  const int wave = tid >> 5;
  const int h    = lane >> 4;
  const int nloc = lane & 15;
  const int mi   = wave >> 1;       // 0..3 -> 32-row slice
  const int ni   = wave & 1;        // 0..1 -> 32-col slice

  // A staging: 2 threads per row, 16 bf16 (32B) each
  const int stRow  = tid >> 1;
  const int stHalf = tid & 1;
  int stEdge = tileBase + stRow;
  if (stEdge >= tCount) stEdge = tCount - 1;            // clamp
  const int srcNode = eidx[bucket[tStart + stEdge]];
  const uint16_t* aSrc = xbf + (size_t)srcNode * KD + stHalf * 16;

  // B staging: 4 threads per row, 8 bf16 (16B) each
  const int bRow = tid >> 2;
  const int bSeg = tid & 3;
  const uint16_t* bSrc = wmsg_bf + ((size_t)t * KD + colBase + bRow) * KD + bSeg * 8;

  uint4* aDst = (uint4*)(sA + stRow * LDSS + stHalf * 16);
  uint4* bDst = (uint4*)(sB + bRow * LDSS + bSeg * 8);

  v8f zero = {0.f,0.f,0.f,0.f,0.f,0.f,0.f,0.f};
  v8f acc[2][2] = {{zero, zero}, {zero, zero}};

  uint4 ra0, ra1, rb;
  {
    const uint4* gp = (const uint4*)aSrc;
    ra0 = gp[0]; ra1 = gp[1];
    rb  = *(const uint4*)bSrc;
  }

  for (int k0 = 0; k0 < KD; k0 += BK) {
    __syncthreads();
    aDst[0] = ra0; aDst[1] = ra1;        // pure copies, no conversion
    *bDst = rb;
    __syncthreads();

    const int kn = k0 + BK;
    if (kn < KD) {
      const uint4* gp = (const uint4*)(aSrc + kn);
      ra0 = gp[0]; ra1 = gp[1];
      rb  = *(const uint4*)(bSrc + kn);
      if (kn + BK < KD) __builtin_prefetch(bSrc + kn + BK, 0, 0);
    }

    // 2 A-frags + 2 B-frags, each consumed by two WMMAs -> all stay live
    const __bf16* pa = sA + (mi * 32 + nloc) * LDSS + 16 * h;
    v16bf a0 = ldfragB(pa);
    v16bf a1 = ldfragB(pa + 16 * LDSS);
    const __bf16* pb = sB + (ni * 32 + nloc) * LDSS + 16 * h;
    v16bf b0 = ldfragB(pb);
    v16bf b1 = ldfragB(pb + 16 * LDSS);

    acc[0][0] = __builtin_amdgcn_wmma_f32_16x16x32_bf16(false, a0, false, b0, (short)0, acc[0][0], false, false);
    acc[0][1] = __builtin_amdgcn_wmma_f32_16x16x32_bf16(false, a0, false, b1, (short)0, acc[0][1], false, false);
    acc[1][0] = __builtin_amdgcn_wmma_f32_16x16x32_bf16(false, a1, false, b0, (short)0, acc[1][0], false, false);
    acc[1][1] = __builtin_amdgcn_wmma_f32_16x16x32_bf16(false, a1, false, b1, (short)0, acc[1][1], false, false);
  }

  // epilogue: + b_msg, scatter-add into agg[tgt]
  float bias[2];
  #pragma unroll
  for (int j = 0; j < 2; ++j) bias[j] = bmsg[t * KD + colBase + ni * 32 + j * 16 + nloc];

  #pragma unroll
  for (int kk = 0; kk < 2; ++kk) {
    for (int r = 0; r < 8; ++r) {
      int mLoc = mi * 32 + kk * 16 + r + 8 * h;
      int erow = tileBase + mLoc;
      if (erow < tCount) {
        int e   = bucket[tStart + erow];
        int tgt = eidx[NEDGES + e];
        #pragma unroll
        for (int j = 0; j < 2; ++j) {
          int col = colBase + ni * 32 + j * 16 + nloc;
          atomicAdd(agg + (size_t)tgt * KD + col, acc[kk][j][r] + bias[j]);
        }
      }
    }
  }
}

// ---------------- GEMM 2: node update (K = 2D, split Xbf | Aggbf) ----------------
// grid: (ceil(N/BM), D/BN), block: 256
__launch_bounds__(256)
__global__ void k_upd_gemm(const float* __restrict__ xf,         // f32 (residual)
                           const uint16_t* __restrict__ xbf,     // [N][D] bf16
                           const uint16_t* __restrict__ aggbf,   // [N][D] bf16
                           const uint16_t* __restrict__ wupd_bf, // [D][2D] bf16
                           const float* __restrict__ bupd,
                           float* __restrict__ out) {
  __shared__ __align__(16) __bf16 sA[BM * LDSS];
  __shared__ __align__(16) __bf16 sB[BN * LDSS];

  const int tileBase = blockIdx.x * BM;
  const int colBase  = blockIdx.y * BN;

  const int tid  = threadIdx.x;
  const int lane = tid & 31;
  const int wave = tid >> 5;
  const int h    = lane >> 4;
  const int nloc = lane & 15;
  const int mi   = wave >> 1;
  const int ni   = wave & 1;

  const int stRow  = tid >> 1;
  const int stHalf = tid & 1;
  int row = tileBase + stRow;
  if (row >= NNODES) row = NNODES - 1;                  // clamp
  const uint16_t* aSrcX = xbf   + (size_t)row * KD + stHalf * 16;
  const uint16_t* aSrcG = aggbf + (size_t)row * KD + stHalf * 16;

  const int bRow = tid >> 2;
  const int bSeg = tid & 3;
  const uint16_t* bSrc = wupd_bf + (size_t)(colBase + bRow) * K2D + bSeg * 8;

  uint4* aDst = (uint4*)(sA + stRow * LDSS + stHalf * 16);
  uint4* bDst = (uint4*)(sB + bRow * LDSS + bSeg * 8);

  v8f zero = {0.f,0.f,0.f,0.f,0.f,0.f,0.f,0.f};
  v8f acc[2][2] = {{zero, zero}, {zero, zero}};

  uint4 ra0, ra1, rb;
  {
    const uint4* gp = (const uint4*)aSrcX;
    ra0 = gp[0]; ra1 = gp[1];
    rb  = *(const uint4*)bSrc;
  }

  for (int k0 = 0; k0 < K2D; k0 += BK) {
    __syncthreads();
    aDst[0] = ra0; aDst[1] = ra1;
    *bDst = rb;
    __syncthreads();

    const int kn = k0 + BK;
    if (kn < K2D) {
      const uint4* gp = (kn < KD) ? (const uint4*)(aSrcX + kn)
                                  : (const uint4*)(aSrcG + (kn - KD));
      ra0 = gp[0]; ra1 = gp[1];
      rb  = *(const uint4*)(bSrc + kn);
      if (kn + BK < K2D) __builtin_prefetch(bSrc + kn + BK, 0, 0);
    }

    const __bf16* pa = sA + (mi * 32 + nloc) * LDSS + 16 * h;
    v16bf a0 = ldfragB(pa);
    v16bf a1 = ldfragB(pa + 16 * LDSS);
    const __bf16* pb = sB + (ni * 32 + nloc) * LDSS + 16 * h;
    v16bf b0 = ldfragB(pb);
    v16bf b1 = ldfragB(pb + 16 * LDSS);

    acc[0][0] = __builtin_amdgcn_wmma_f32_16x16x32_bf16(false, a0, false, b0, (short)0, acc[0][0], false, false);
    acc[0][1] = __builtin_amdgcn_wmma_f32_16x16x32_bf16(false, a0, false, b1, (short)0, acc[0][1], false, false);
    acc[1][0] = __builtin_amdgcn_wmma_f32_16x16x32_bf16(false, a1, false, b0, (short)0, acc[1][0], false, false);
    acc[1][1] = __builtin_amdgcn_wmma_f32_16x16x32_bf16(false, a1, false, b1, (short)0, acc[1][1], false, false);
  }

  // epilogue: bias + ReLU + residual, pre-LN value into d_out
  float bias[2];
  #pragma unroll
  for (int j = 0; j < 2; ++j) bias[j] = bupd[colBase + ni * 32 + j * 16 + nloc];

  #pragma unroll
  for (int kk = 0; kk < 2; ++kk) {
    for (int r = 0; r < 8; ++r) {
      int mLoc = mi * 32 + kk * 16 + r + 8 * h;
      int orow = tileBase + mLoc;
      if (orow < NNODES) {
        #pragma unroll
        for (int j = 0; j < 2; ++j) {
          int col = colBase + ni * 32 + j * 16 + nloc;
          float u = fmaxf(acc[kk][j][r] + bias[j], 0.f);
          out[(size_t)orow * KD + col] = xf[(size_t)orow * KD + col] + u;
        }
      }
    }
  }
}

// ---------------- LayerNorm in place on d_out ----------------
__launch_bounds__(256)
__global__ void k_layernorm(float* __restrict__ out,
                            const float* __restrict__ gamma,
                            const float* __restrict__ beta) {
  const int row  = blockIdx.x;
  const int tid  = threadIdx.x;
  const int lane = tid & 31;
  const int wave = tid >> 5;
  const size_t base = (size_t)row * KD;

  float x0 = out[base + tid];
  float x1 = out[base + tid + 256];
  float x2 = out[base + tid + 512];
  float s = x0 + x1 + x2;
  float q = x0 * x0 + x1 * x1 + x2 * x2;
  #pragma unroll
  for (int off = 16; off > 0; off >>= 1) {
    s += __shfl_down(s, off);
    q += __shfl_down(q, off);
  }
  __shared__ float ss[8], sq[8];
  if (lane == 0) { ss[wave] = s; sq[wave] = q; }
  __syncthreads();
  float S = 0.f, Q = 0.f;
  #pragma unroll
  for (int w = 0; w < 8; ++w) { S += ss[w]; Q += sq[w]; }
  const float mu  = S * (1.0f / KD);
  const float var = Q * (1.0f / KD) - mu * mu;
  const float rs  = rsqrtf(var + 1e-5f);
  out[base + tid]       = (x0 - mu) * rs * gamma[tid]       + beta[tid];
  out[base + tid + 256] = (x1 - mu) * rs * gamma[tid + 256] + beta[tid + 256];
  out[base + tid + 512] = (x2 - mu) * rs * gamma[tid + 512] + beta[tid + 512];
}

// ---------------- launch ----------------
extern "C" void kernel_launch(void* const* d_in, const int* in_sizes, int n_in,
                              void* d_out, int out_size, void* d_ws, size_t ws_size,
                              hipStream_t stream) {
  const float* xf    = (const float*)d_in[0];
  const float* wmsg  = (const float*)d_in[1];
  const float* bmsg  = (const float*)d_in[2];
  const float* wupd  = (const float*)d_in[3];
  const float* bupd  = (const float*)d_in[4];
  const float* gamma = (const float*)d_in[5];
  const float* beta  = (const float*)d_in[6];
  const int*   eidx  = (const int*)d_in[7];
  const int*   etype = (const int*)d_in[8];
  float* out = (float*)d_out;

  // workspace layout (256B aligned regions)
  char* ws = (char*)d_ws;
  size_t off = 0;
  uint16_t* xbf     = (uint16_t*)(ws + off); off += (size_t)NNODES * KD * 2;        // 153.6 MB
  uint16_t* wmsg_bf = (uint16_t*)(ws + off); off += (size_t)NTYPES * KD * KD * 2;   // 5.9 MB
  uint16_t* wupd_bf = (uint16_t*)(ws + off); off += (size_t)KD * K2D * 2;           // 2.36 MB
  uint16_t* aggbf   = (uint16_t*)(ws + off); off += (size_t)NNODES * KD * 2;        // 153.6 MB
  int* meta         = (int*)(ws + off);      off += 256;
  int* bucket       = (int*)(ws + off);      off += (size_t)NEDGES * sizeof(int);
  off = (off + 255) & ~(size_t)255;
  float* agg        = (float*)(ws + off);    // N*D f32 = 307.2 MB

  // one-time precision conversions (bf16 feeds the WMMA pipelines)
  k_f32_to_bf16<<<4096, 256, 0, stream>>>((const float2*)xf,   (uint32_t*)xbf,
                                          (int64_t)NNODES * KD / 2);
  k_f32_to_bf16<<<2048, 256, 0, stream>>>((const float2*)wmsg, (uint32_t*)wmsg_bf,
                                          (int64_t)NTYPES * KD * KD / 2);
  k_f32_to_bf16<<<1024, 256, 0, stream>>>((const float2*)wupd, (uint32_t*)wupd_bf,
                                          (int64_t)KD * K2D / 2);
  k_zero<<<4096, 256, 0, stream>>>((float4*)agg, meta);

  // bucket edges by type
  k_hist<<<(NEDGES + 255) / 256, 256, 0, stream>>>(etype, meta);
  k_offsets<<<1, 32, 0, stream>>>(meta);
  k_scatter<<<(NEDGES + 255) / 256, 256, 0, stream>>>(etype, meta, bucket);

  // per-type message GEMM + scatter-add
  dim3 g1((NEDGES + BM - 1) / BM, KD / BN, NTYPES);
  k_msg_gemm<<<g1, 256, 0, stream>>>(xbf, wmsg_bf, bmsg, eidx, meta, bucket, agg);

  // aggregated -> bf16 for the update GEMM's K-second-half
  k_f32_to_bf16<<<4096, 256, 0, stream>>>((const float2*)agg, (uint32_t*)aggbf,
                                          (int64_t)NNODES * KD / 2);

  // node update GEMM (bias + ReLU + residual fused)
  dim3 g2((NNODES + BM - 1) / BM, KD / BN);
  k_upd_gemm<<<g2, 256, 0, stream>>>(xf, xbf, aggbf, wupd_bf, bupd, out);

  k_layernorm<<<NNODES, 256, 0, stream>>>(out, gamma, beta);
}
